// GCN_8014408974455
// MI455X (gfx1250) — matrix operand order repaired and checked by reading, not compiled, not run on hardware
//
#include <hip/hip_runtime.h>
#include <hip/hip_bf16.h>
#include <math.h>

// ---------------------------------------------------------------------------
// GCN + VQ pipeline for MI455X (gfx1250, wave32, WMMA).
// GEMM path: all operands pre-converted to f16 (weights pre-transposed), so
// every fragment load is a contiguous 16B global_load_b128 matching the ISA
// wave32 WMMA layouts. Each wave computes a 32x32 output tile: 4 v_wmma per
// k-step with 2x operand reuse. adj_q = qe@qe^T is never materialized: the
// WMMA tile sweep accumulates (min, max, sum, sumsq) and the sparse-adjacency
// correction is a per-edge dot product.
// ---------------------------------------------------------------------------

typedef __attribute__((ext_vector_type(16))) _Float16 v16h;
typedef __attribute__((ext_vector_type(8)))  _Float16 v8h;
typedef __attribute__((ext_vector_type(8)))  float    v8f;

// A-fragment (16x32 f16, ISA 7.12.2): lane -> M, per-lane K runs are
// [half*8, half*8+8) and [16+half*8, 16+half*8+8): two b128 loads.
__device__ inline v16h load_afrag(const _Float16* __restrict__ p)
{
    v8h lo = *(const v8h*)(p);
    v8h hi = *(const v8h*)(p + 16);
    return __builtin_shufflevector(lo, hi, 0,1,2,3,4,5,6,7,8,9,10,11,12,13,14,15);
}

// B-fragment (32x16 f16) from B^T storage [Ncol,K]: lane -> N, per-lane K runs
// are [half*8, half*8+8) and [16+half*8, 16+half*8+8): two b128 loads.
__device__ inline v16h load_bfrag(const _Float16* __restrict__ p)
{
    v8h lo = *(const v8h*)(p);
    v8h hi = *(const v8h*)(p + 16);
    return __builtin_shufflevector(lo, hi, 0,1,2,3,4,5,6,7,8,9,10,11,12,13,14,15);
}

__device__ inline v8f wmma_f16(v16h a, v16h b, v8f c)
{
    return __builtin_amdgcn_wmma_f32_16x16x32_f16(
        /*neg_a=*/false, a, /*neg_b=*/false, b,
        /*c_mod=*/(short)0, c, /*reuse_a=*/false, /*reuse_b=*/false);
}

// ---------------- 32x32-tile GEMM: C = act(A @ Bt^T (+bias)) ----------------
// A: f16 [M,K] row-major.  Bt: f16 [Ncol,K] row-major.  C: f32 [M,Ncol].
template<bool RELU, bool BIAS>
__global__ void gemm32_kernel(const _Float16* __restrict__ A,
                              const _Float16* __restrict__ Bt,
                              const float* __restrict__ bias,
                              float* __restrict__ C,
                              int M, int Ncol, int K)
{
    const int lane = threadIdx.x & 31;
    const int nr   = lane & 15;
    const int half = lane >> 4;
    const int col0 = blockIdx.x * 32;
    const int row0 = blockIdx.y * 32;

    const _Float16* a0p = A  + (size_t)(row0 + nr) * K + half * 8;
    const _Float16* a1p = a0p + (size_t)16 * K;
    const _Float16* b0p = Bt + (size_t)(col0 + nr) * K + half * 8;
    const _Float16* b1p = b0p + (size_t)16 * K;

    v8f acc00 = {}, acc01 = {}, acc10 = {}, acc11 = {};
    for (int k0 = 0; k0 < K; k0 += 32) {
        v16h a0 = load_afrag(a0p + k0);
        v16h a1 = load_afrag(a1p + k0);
        v16h b0 = load_bfrag(b0p + k0);
        v16h b1 = load_bfrag(b1p + k0);
        acc00 = wmma_f16(a0, b0, acc00);
        acc01 = wmma_f16(a0, b1, acc01);
        acc10 = wmma_f16(a1, b0, acc10);
        acc11 = wmma_f16(a1, b1, acc11);
    }

    const int mh = half * 8;   // C/D layout: VGPR r -> M = r + half*8
    auto store_tile = [&](v8f acc, int dr, int dc) {
        const float bv = BIAS ? bias[col0 + dc + nr] : 0.0f;
#pragma unroll
        for (int r = 0; r < 8; ++r) {
            float v = acc[r] + bv;
            if (RELU) v = fmaxf(v, 0.0f);
            C[(size_t)(row0 + dr + mh + r) * Ncol + col0 + dc + nr] = v;
        }
    };
    store_tile(acc00, 0, 0);
    store_tile(acc01, 0, 16);
    store_tile(acc10, 16, 0);
    store_tile(acc11, 16, 16);
}

// ---------------- float atomic min/max via CAS -------------------------------
__device__ inline void atomicMinF(float* addr, float val) {
    unsigned* ua = (unsigned*)addr;
    unsigned old = __float_as_uint(*addr);
    while (__uint_as_float(old) > val) {
        unsigned assumed = old;
        old = atomicCAS(ua, assumed, __float_as_uint(val));
        if (old == assumed) break;
    }
}
__device__ inline void atomicMaxF(float* addr, float val) {
    unsigned* ua = (unsigned*)addr;
    unsigned old = __float_as_uint(*addr);
    while (__uint_as_float(old) < val) {
        unsigned assumed = old;
        old = atomicCAS(ua, assumed, __float_as_uint(val));
        if (old == assumed) break;
    }
}

// ---------------- adj_q statistics sweep (min,max,sum,sumsq) -----------------
// One wave per 32x32 tile of qe @ qe^T; nothing materialized.
__global__ void adj_stats32_kernel(const _Float16* __restrict__ qeh, float* sc, int K)
{
    const int lane = threadIdx.x & 31;
    const int nr   = lane & 15;
    const int half = lane >> 4;
    const int col0 = blockIdx.x * 32;
    const int row0 = blockIdx.y * 32;

    const _Float16* a0p = qeh + (size_t)(row0 + nr) * K + half * 8;
    const _Float16* a1p = a0p + (size_t)16 * K;
    const _Float16* b0p = qeh + (size_t)(col0 + nr) * K + half * 8;
    const _Float16* b1p = b0p + (size_t)16 * K;

    v8f acc00 = {}, acc01 = {}, acc10 = {}, acc11 = {};
    for (int k0 = 0; k0 < K; k0 += 32) {
        v16h a0 = load_afrag(a0p + k0);
        v16h a1 = load_afrag(a1p + k0);
        v16h b0 = load_bfrag(b0p + k0);
        v16h b1 = load_bfrag(b1p + k0);
        acc00 = wmma_f16(a0, b0, acc00);
        acc01 = wmma_f16(a0, b1, acc01);
        acc10 = wmma_f16(a1, b0, acc10);
        acc11 = wmma_f16(a1, b1, acc11);
    }

    float lmin = acc00[0], lmax = acc00[0], s1 = 0.0f, s2 = 0.0f;
    auto fold = [&](v8f acc) {
#pragma unroll
        for (int r = 0; r < 8; ++r) {
            float v = acc[r];
            lmin = fminf(lmin, v); lmax = fmaxf(lmax, v);
            s1 += v; s2 += v * v;
        }
    };
    fold(acc00); fold(acc01); fold(acc10); fold(acc11);
#pragma unroll
    for (int off = 16; off; off >>= 1) {
        lmin = fminf(lmin, __shfl_xor(lmin, off, 32));
        lmax = fmaxf(lmax, __shfl_xor(lmax, off, 32));
        s1 += __shfl_xor(s1, off, 32);
        s2 += __shfl_xor(s2, off, 32);
    }
    if (lane == 0) {
        atomicMinF(&sc[0], lmin);
        atomicMaxF(&sc[1], lmax);
        atomicAdd(&sc[2], s1);
        atomicAdd(&sc[3], s2);
    }
}

// ---------------- conversion helpers ----------------------------------------
__global__ void convert_kernel(const float* __restrict__ x, _Float16* __restrict__ y, int n)
{
    int i = blockIdx.x * blockDim.x + threadIdx.x;
    if (i < n) y[i] = (_Float16)x[i];
}

// B [K,Ncol] f32 row-major -> Bt [Ncol,K] f16 row-major
__global__ void convert_T_kernel(const float* __restrict__ B, _Float16* __restrict__ Bt,
                                 int K, int Ncol)
{
    int i = blockIdx.x * blockDim.x + threadIdx.x;
    if (i < K * Ncol) {
        int k = i / Ncol, n = i - k * Ncol;
        Bt[(size_t)n * K + k] = (_Float16)B[i];
    }
}

// ---------------- graph-conv helpers ----------------------------------------
__global__ void degree_kernel(const int* __restrict__ src, const int* __restrict__ dst,
                              float* degOut, float* degIn, int E)
{
    int e = blockIdx.x * blockDim.x + threadIdx.x;
    if (e < E) {
        atomicAdd(&degOut[src[e]], 1.0f);
        atomicAdd(&degIn[dst[e]], 1.0f);
    }
}

__global__ void rsqrt_kernel(const float* __restrict__ deg, float* out, int n)
{
    int i = blockIdx.x * blockDim.x + threadIdx.x;
    if (i < n) {
        float d = deg[i];
        out[i] = d > 0.0f ? rsqrtf(fmaxf(d, 1e-12f)) : 0.0f;
    }
}

__global__ void rowscale_kernel(const float* __restrict__ x, const float* __restrict__ s,
                                float* __restrict__ y, int total, int D)
{
    int i = blockIdx.x * blockDim.x + threadIdx.x;
    if (i < total) y[i] = x[i] * s[i / D];
}

// one block (= D threads) per edge: m[dst] += hsc[src]
__global__ void scatter_kernel(const float* __restrict__ hsc, const int* __restrict__ src,
                               const int* __restrict__ dst, float* __restrict__ m, int D)
{
    const int e = blockIdx.x;
    const int d = threadIdx.x;
    atomicAdd(&m[(size_t)dst[e] * D + d], hsc[(size_t)src[e] * D + d]);
}

// ---------------- row L2-normalize (one wave per row), dual f32/f16 out ------
__global__ void rownorm_kernel(const float* __restrict__ x, float* __restrict__ y,
                               _Float16* __restrict__ yh, int D)
{
    const int row  = blockIdx.x * (blockDim.x >> 5) + (threadIdx.x >> 5);
    const int lane = threadIdx.x & 31;
    const float* xr = x + (size_t)row * D;
    float ss = 0.0f;
    for (int d = lane; d < D; d += 32) { float v = xr[d]; ss += v * v; }
#pragma unroll
    for (int off = 16; off; off >>= 1) ss += __shfl_xor(ss, off, 32);
    const float rinv = 1.0f / fmaxf(sqrtf(ss), 1e-12f);
    for (int d = lane; d < D; d += 32) {
        float v = xr[d] * rinv;
        if (y)  y[(size_t)row * D + d]  = v;
        if (yh) yh[(size_t)row * D + d] = (_Float16)v;
    }
}

// ---------------- argmax over dist rows (one wave per row) -------------------
__global__ void argmax_kernel(const float* __restrict__ dist, int* __restrict__ ind, int C)
{
    const int row  = blockIdx.x * (blockDim.x >> 5) + (threadIdx.x >> 5);
    const int lane = threadIdx.x & 31;
    const float* r = dist + (size_t)row * C;
    float best = -__builtin_inff(); int bidx = 0x7FFFFFFF;
    for (int c = lane; c < C; c += 32) {
        float v = r[c];
        if (v > best) { best = v; bidx = c; }
    }
#pragma unroll
    for (int off = 16; off; off >>= 1) {
        float ov = __shfl_xor(best, off, 32);
        int   oi = __shfl_xor(bidx, off, 32);
        if (ov > best || (ov == best && oi < bidx)) { best = ov; bidx = oi; }
    }
    if (lane == 0) ind[row] = bidx;
}

// quantized = cn[ind] (f32 + f16 copies); accumulate commit sum (q - h1)^2
__global__ void gather_commit_kernel(const float* __restrict__ cn, const int* __restrict__ ind,
                                     const float* __restrict__ h1, float* __restrict__ quant,
                                     _Float16* __restrict__ quanth, float* commitAcc, int D)
{
    const int row  = blockIdx.x * (blockDim.x >> 5) + (threadIdx.x >> 5);
    const int lane = threadIdx.x & 31;
    const float* qr = cn + (size_t)ind[row] * D;
    const float* hr = h1 + (size_t)row * D;
    float s = 0.0f;
    for (int d = lane; d < D; d += 32) {
        float q = qr[d];
        quant [(size_t)row * D + d] = q;
        quanth[(size_t)row * D + d] = (_Float16)q;
        float dd = q - hr[d];
        s += dd * dd;
    }
#pragma unroll
    for (int off = 16; off; off >>= 1) s += __shfl_xor(s, off, 32);
    if (lane == 0) atomicAdd(commitAcc, s);
}

// sum (a-b)^2 with grid-stride + wave reduce
__global__ void sqdiff_kernel(const float* __restrict__ a, const float* __restrict__ b,
                              float* acc, int n)
{
    float s = 0.0f;
    for (int i = blockIdx.x * blockDim.x + threadIdx.x; i < n; i += gridDim.x * blockDim.x) {
        float d = a[i] - b[i];
        s += d * d;
    }
#pragma unroll
    for (int off = 16; off; off >>= 1) s += __shfl_xor(s, off, 32);
    if ((threadIdx.x & 31) == 0) atomicAdd(acc, s);
}

// per-edge: a = dot(qe[src], qe[dst]); acc += 1 - 2*(a-mn)/(mx-mn)
__global__ void edge_term_kernel(const float* __restrict__ qe, const int* __restrict__ src,
                                 const int* __restrict__ dst, const float* __restrict__ sc,
                                 float* acc, int D)
{
    const int e    = blockIdx.x * (blockDim.x >> 5) + (threadIdx.x >> 5);
    const int lane = threadIdx.x & 31;
    const float* a = qe + (size_t)src[e] * D;
    const float* b = qe + (size_t)dst[e] * D;
    float s = 0.0f;
    for (int d = lane; d < D; d += 32) s += a[d] * b[d];
#pragma unroll
    for (int off = 16; off; off >>= 1) s += __shfl_xor(s, off, 32);
    if (lane == 0) {
        float mn = sc[0], mx = sc[1];
        float an = (s - mn) / (mx - mn);
        atomicAdd(acc, 1.0f - 2.0f * an);
    }
}

__global__ void init_scalars_kernel(float* sc)
{
    int t = threadIdx.x;
    if (t < 16) sc[t] = 0.0f;
    if (t == 0) sc[0] =  __builtin_inff();
    if (t == 1) sc[1] = -__builtin_inff();
}

// loss = node_rec + edge_rec + commit (closed form over streamed stats)
__global__ void finalize_kernel(const float* __restrict__ sc, float* lossOut, int N, int D)
{
    if (threadIdx.x == 0 && blockIdx.x == 0) {
        double mn = sc[0], mx = sc[1], S1 = sc[2], S2 = sc[3];
        double eS = sc[4], cS = sc[5], nS = sc[6];
        const double NN = (double)N * (double)N;
        const double ND = (double)N * (double)D;
        double r    = mx - mn;
        double inv2 = 1.0 / (r * r);
        double sumSq = (S2 - 2.0 * mn * S1 + NN * mn * mn) * inv2;  // sum a_norm^2
        double edge  = sqrt((sumSq + eS) / NN);                     // LAMB_EDGE = 1
        double node  = nS / ND;                                     // LAMB_NODE = 1
        double commit = 0.25 * cS / ND;                             // COMMIT_W
        lossOut[0] = (float)(node + edge + commit);
    }
}

// ---------------------------------------------------------------------------
extern "C" void kernel_launch(void* const* d_in, const int* in_sizes, int n_in,
                              void* d_out, int out_size, void* d_ws, size_t ws_size,
                              hipStream_t stream)
{
    const int D = 128, H = 128, O = 64;
    const int N = in_sizes[0] / D;        // 8192
    const int E = in_sizes[1];            // 262144
    const int C = in_sizes[13] / D;       // 4096

    const float* feats   = (const float*)d_in[0];
    const int*   src     = (const int*)  d_in[1];
    const int*   dst     = (const int*)  d_in[2];
    const float* W1      = (const float*)d_in[3];
    const float* b1      = (const float*)d_in[4];
    const float* W2      = (const float*)d_in[5];
    const float* b2      = (const float*)d_in[6];
    const float* dec1_W  = (const float*)d_in[7];
    const float* dec1_b  = (const float*)d_in[8];
    const float* dec2_W  = (const float*)d_in[9];
    const float* dec2_b  = (const float*)d_in[10];
    const float* lin_W   = (const float*)d_in[11];
    const float* lin_b   = (const float*)d_in[12];
    const float* cb      = (const float*)d_in[13];

    // ---- d_out layout (floats): h1, quantized, h3, out, loss, dist, cn ----
    float* out_f   = (float*)d_out;
    float* h1      = out_f;
    float* quant   = h1    + (size_t)N * D;
    float* h3      = quant + (size_t)N * D;
    float* head    = h3    + (size_t)N * H;
    float* lossPtr = head  + (size_t)N * O;
    float* dist    = lossPtr + 1;
    float* cn      = dist  + (size_t)N * C;

    // ---- workspace carve ----
    float* ws      = (float*)d_ws;
    float* degOut  = ws;                         // N
    float* degIn   = degOut + N;                 // N
    float* normSrc = degIn  + N;                 // N
    float* normDst = normSrc + N;                // N
    float* hsc     = normDst + N;                // N*D
    float* magg    = hsc  + (size_t)N * D;       // N*D
    float* qe      = magg + (size_t)N * D;       // N*D
    float* qn      = qe   + (size_t)N * D;       // N*D
    int*   ind     = (int*)(qn + (size_t)N * D); // N ints
    float* sc      = (float*)(ind + N);          // 16 scalars
    // sc: [0]=min [1]=max [2]=S1 [3]=S2 [4]=edgeSum [5]=commitSum [6]=nodeSum

    // f16 region (32B aligned for b128 fragment loads)
    uintptr_t hp = ((uintptr_t)(sc + 16) + 31) & ~(uintptr_t)31;
    _Float16* agg_h   = (_Float16*)hp;                 // N*D
    _Float16* xn_h    = agg_h   + (size_t)N * D;       // N*D
    _Float16* cn_h    = xn_h    + (size_t)N * D;       // C*D
    _Float16* quant_h = cn_h    + (size_t)C * D;       // N*D
    _Float16* qe_h    = quant_h + (size_t)N * D;       // N*D
    _Float16* h3_h    = qe_h    + (size_t)N * D;       // N*H
    _Float16* W1t     = h3_h    + (size_t)N * H;       // D*D
    _Float16* W2t     = W1t + (size_t)D * D;           // D*H
    _Float16* d1t     = W2t + (size_t)D * H;           // D*D
    _Float16* d2t     = d1t + (size_t)D * D;           // D*D
    _Float16* lint    = d2t + (size_t)D * D;           // H*O

    const dim3 b32(32), b128(128), b256(256);

    // ---- init + one-shot weight converts (transposed f16) ----
    hipMemsetAsync(degOut, 0, (size_t)2 * N * sizeof(float), stream);
    init_scalars_kernel<<<1, 32, 0, stream>>>(sc);
    convert_T_kernel<<<(D * D + 255) / 256, b256, 0, stream>>>(W1, W1t, D, D);
    convert_T_kernel<<<(D * H + 255) / 256, b256, 0, stream>>>(W2, W2t, D, H);
    convert_T_kernel<<<(D * D + 255) / 256, b256, 0, stream>>>(dec1_W, d1t, D, D);
    convert_T_kernel<<<(D * D + 255) / 256, b256, 0, stream>>>(dec2_W, d2t, D, D);
    convert_T_kernel<<<(H * O + 255) / 256, b256, 0, stream>>>(lin_W, lint, H, O);

    // ---- degrees & norms ----
    degree_kernel<<<(E + 255) / 256, b256, 0, stream>>>(src, dst, degOut, degIn, E);
    rsqrt_kernel<<<(N + 255) / 256, b256, 0, stream>>>(degOut, normSrc, N);
    rsqrt_kernel<<<(N + 255) / 256, b256, 0, stream>>>(degIn, normDst, N);

    // ---- layer 1 graph conv: h1 = relu((D^-1/2 A D^-1/2 X) W1 + b1) ----
    rowscale_kernel<<<(N * D + 255) / 256, b256, 0, stream>>>(feats, normSrc, hsc, N * D, D);
    hipMemsetAsync(magg, 0, (size_t)N * D * sizeof(float), stream);
    scatter_kernel<<<E, dim3(D), 0, stream>>>(hsc, src, dst, magg, D);
    rowscale_kernel<<<(N * D + 255) / 256, b256, 0, stream>>>(magg, normDst, magg, N * D, D);
    convert_kernel<<<(N * D + 255) / 256, b256, 0, stream>>>(magg, agg_h, N * D);
    gemm32_kernel<true, true><<<dim3(D / 32, N / 32), b32, 0, stream>>>(
        agg_h, W1t, b1, h1, N, D, D);

    // ---- VQ: cosine codebook ----
    rownorm_kernel<<<N / 4, b128, 0, stream>>>(h1, nullptr, xn_h, D);
    rownorm_kernel<<<C / 4, b128, 0, stream>>>(cb, cn, cn_h, D);
    gemm32_kernel<false, false><<<dim3(C / 32, N / 32), b32, 0, stream>>>(
        xn_h, cn_h, nullptr, dist, N, C, D);
    argmax_kernel<<<N / 4, b128, 0, stream>>>(dist, ind, C);
    gather_commit_kernel<<<N / 4, b128, 0, stream>>>(cn, ind, h1, quant, quant_h, &sc[5], D);

    // ---- decoders ----
    gemm32_kernel<false, true><<<dim3(D / 32, N / 32), b32, 0, stream>>>(
        quant_h, d1t, dec1_b, qe, N, D, D);
    gemm32_kernel<false, true><<<dim3(D / 32, N / 32), b32, 0, stream>>>(
        quant_h, d2t, dec2_b, qn, N, D, D);
    sqdiff_kernel<<<2048, b256, 0, stream>>>(h1, qn, &sc[6], N * D);

    // ---- adjacency reconstruction loss (streamed, no NxN buffer) ----
    convert_kernel<<<(N * D + 255) / 256, b256, 0, stream>>>(qe, qe_h, N * D);
    adj_stats32_kernel<<<dim3(N / 32, N / 32), b32, 0, stream>>>(qe_h, sc, D);
    edge_term_kernel<<<E / 4, b128, 0, stream>>>(qe, src, dst, sc, &sc[4], D);

    // ---- layer 2 graph conv on qe ----
    rowscale_kernel<<<(N * D + 255) / 256, b256, 0, stream>>>(qe, normSrc, hsc, N * D, D);
    hipMemsetAsync(magg, 0, (size_t)N * D * sizeof(float), stream);
    scatter_kernel<<<E, dim3(D), 0, stream>>>(hsc, src, dst, magg, D);
    rowscale_kernel<<<(N * D + 255) / 256, b256, 0, stream>>>(magg, normDst, magg, N * D, D);
    convert_kernel<<<(N * D + 255) / 256, b256, 0, stream>>>(magg, agg_h, N * D);
    gemm32_kernel<true, true><<<dim3(H / 32, N / 32), b32, 0, stream>>>(
        agg_h, W2t, b2, h3, N, H, D);

    // ---- linear head ----
    convert_kernel<<<(N * H + 255) / 256, b256, 0, stream>>>(h3, h3_h, N * H);
    gemm32_kernel<false, true><<<dim3(O / 32, N / 32), b32, 0, stream>>>(
        h3_h, lint, lin_b, head, N, O, H);

    // ---- total loss ----
    finalize_kernel<<<1, 32, 0, stream>>>(sc, lossPtr, N, D);
}
